// AdvancedPositionSelector_26448408609234
// MI455X (gfx1250) — matrix operand order repaired and checked by reading, not compiled, not run on hardware
//
#include <hip/hip_runtime.h>
#include <hip/hip_bf16.h>

#define Bq   16
#define Sq   8192
#define Dq   256
#define D2q  128
#define Kq   2048
#define EPSq 1e-5f

typedef __attribute__((ext_vector_type(16))) __bf16 v16bf;
typedef __attribute__((ext_vector_type(8)))  float  v8f;
typedef __attribute__((ext_vector_type(4)))  unsigned int u32x4;
typedef __attribute__((ext_vector_type(8)))  int          i32x8;
typedef __attribute__((ext_vector_type(4)))  int          i32x4;

// ---------------- d_out layout (floats) ----------------
// selected [16,2048,256] | top_indices [16,2048] | final_scores [16,8192] | current_gw [1]
#define OUT_SEL 0
#define OUT_IDX (Bq*Kq*Dq)
#define OUT_SC  (OUT_IDX + Bq*Kq)
#define OUT_GW  (OUT_SC + Bq*Sq)

// ---------------- workspace layout (4-byte units) ----------------
#define WS_W1P  0        // 32768 u32 : W1 bf16 pairs in WMMA-B layout [kk8][nt16][lane32][v8]
#define WS_W2P  32768    // 16384 u32 : BN-folded W2 bf16 pairs [kk8][nt8][lane32][v8]
#define WS_B1   49152    // 256 f32
#define WS_B2   49408    // 128 f32 (BN-folded bias)
#define WS_W3   49536    // 128 f32
#define WS_TV   49664    // 8192 f32 : gw' * enhanced_teacher
#define WS_SCAL 57856    // 4 f32 : [0]=1-gw', [1]=b3
#define WS_TIDX 57860    // 32768 int : top indices

__device__ __forceinline__ unsigned short f2bf(float x) {
    unsigned int u = __float_as_uint(x);
    return (unsigned short)((u + 0x7FFFu + ((u >> 16) & 1u)) >> 16); // RNE
}

// ---------------------------------------------------------------------------
// TDM: 1-D contiguous tile, data_size = 8B units, per CDNA5 ISA §8.3/8.4.
//  g0: [1:0]=count=1 | lds_addr[63:32] | global_addr[120:64] | type[127:126]=2
//  g1: data_size[17:16]=3, tensor_dim0[79:48], tensor_dim1[111:80]=1,
//      tile_dim0[127:112], tile_dim1[143:128]=1, tensor_dim0_stride[207:160]
// This toolchain exposes the 6-arg builtin (g0, g1, g2, g3, g_extra, cpol).
// ---------------------------------------------------------------------------
__device__ __forceinline__ void tdm_load_1d(const void* gptr, unsigned lds_off,
                                            unsigned tile_u8, unsigned tensor_u8) {
    unsigned long long ga = (unsigned long long)gptr;
    u32x4 g0;
    g0.x = 1u;                                      // count=1, user mode
    g0.y = lds_off;                                 // LDS byte address
    g0.z = (unsigned)ga;                            // global_addr[31:0]
    g0.w = (unsigned)(ga >> 32) | (2u << 30);       // global_addr[56:32] | type=2
    i32x8 g1 = {};
    g1[0] = (int)(3u << 16);                        // data_size = 8 bytes
    g1[1] = (int)((tensor_u8 & 0xFFFFu) << 16);     // tensor_dim0[15:0]
    g1[2] = (int)(((tensor_u8 >> 16) & 0xFFFFu) | (1u << 16)); // dim0[31:16], tensor_dim1=1
    g1[3] = (int)(tile_u8 << 16);                   // tile_dim0
    g1[4] = 1;                                      // tile_dim1 = 1
    g1[5] = (int)tensor_u8;                         // tensor_dim0_stride[31:0]
    g1[6] = 0;
    g1[7] = 0;
    i32x4 z4 = {};
    i32x8 z8 = {};
    __builtin_amdgcn_tensor_load_to_lds(g0, g1, z4, z4, z8, 0);
}

__device__ __forceinline__ unsigned lds_off_of(const void* p) {
    return (unsigned)(unsigned long long)p;         // generic LDS addr: low 32 bits = offset
}

// ---------------------------------------------------------------------------
// Prep: pack W1 and BN-folded W2 into the WMMA B-operand lane layout (bf16).
// B operand (bf16 32x16): lane n -> col = n&15, VGPR v -> K = kk*32+8*(n>>4)+2*(v&3)+16*(v>>2).
// ---------------------------------------------------------------------------
__global__ void prep_weights(const float* __restrict__ W1, const float* __restrict__ W2,
                             const float* __restrict__ gamma, const float* __restrict__ var,
                             unsigned int* __restrict__ ws) {
    int tid = blockIdx.x * blockDim.x + threadIdx.x;
    if (tid < 32768) {                 // W1 pack: [kk8][nt16][lane32][v8]
        int v = tid & 7, lane = (tid >> 3) & 31, nt = (tid >> 8) & 15, kk = tid >> 12;
        int col = nt * 16 + (lane & 15);
        int k0  = kk * 32 + 8 * (lane >> 4) + 2 * (v & 3) + 16 * (v >> 2);
        unsigned int lo = f2bf(W1[k0 * Dq + col]);
        unsigned int hi = f2bf(W1[(k0 + 1) * Dq + col]);
        ws[WS_W1P + tid] = lo | (hi << 16);
    } else {                           // W2' pack: [kk8][nt8][lane32][v8]
        int t = tid - 32768;
        int v = t & 7, lane = (t >> 3) & 31, nt = (t >> 8) & 7, kk = t >> 11;
        int col = nt * 16 + (lane & 15);
        int k0  = kk * 32 + 8 * (lane >> 4) + 2 * (v & 3) + 16 * (v >> 2);
        float a0 = gamma[k0]     * rsqrtf(var[k0]     + EPSq);
        float a1 = gamma[k0 + 1] * rsqrtf(var[k0 + 1] + EPSq);
        unsigned int lo = f2bf(a0 * W2[k0 * D2q + col]);
        unsigned int hi = f2bf(a1 * W2[(k0 + 1) * D2q + col]);
        ws[WS_W2P + t] = lo | (hi << 16);
    }
}

__global__ void prep_vec(const float* __restrict__ b1, const float* __restrict__ gamma,
                         const float* __restrict__ beta, const float* __restrict__ mean,
                         const float* __restrict__ var, const float* __restrict__ W2,
                         const float* __restrict__ b2, const float* __restrict__ W3,
                         const float* __restrict__ b3, const float* __restrict__ tg,
                         const float* __restrict__ tc, const float* __restrict__ gw,
                         const float* __restrict__ cw, const float* __restrict__ tp,
                         float* __restrict__ wsf, float* __restrict__ out) {
    int t = blockIdx.x * blockDim.x + threadIdx.x;    // 0..8191
    float gwp = gw[0] * powf(0.95f, tp[0] * 15.0f);
    float g   = tg[t];
    wsf[WS_TV + t] = gwp * (g + cw[0] * tc[t] * g);
    if (t < Dq) wsf[WS_B1 + t] = b1[t];
    if (t < D2q) {
        float acc = b2[t];
        for (int d = 0; d < Dq; ++d) {
            float a = gamma[d] * rsqrtf(var[d] + EPSq);
            float c = beta[d] - mean[d] * a;
            acc += c * W2[d * D2q + t];
        }
        wsf[WS_B2 + t] = acc;
        wsf[WS_W3 + t] = W3[t];
    }
    if (t == 0) {
        wsf[WS_SCAL + 0] = 1.0f - gwp;
        wsf[WS_SCAL + 1] = b3[0];
        out[OUT_GW]      = gwp;
    }
}

// ---------------------------------------------------------------------------
// Fused MLP scores: 32 positions/block, 8 waves, WMMA bf16->f32.
// Weights streamed into LDS by the Tensor Data Mover, double-buffered per
// K-slice and overlapped with WMMA compute (TENSORcnt + barrier pipeline).
// ---------------------------------------------------------------------------
__global__ __launch_bounds__(256) void scores_kernel(const float* __restrict__ emb,
                                                     const unsigned int* __restrict__ wsu,
                                                     const float* __restrict__ wsf,
                                                     float* __restrict__ out) {
    __shared__ __align__(16) unsigned char xh_raw[16896]; // xs bf16[32][264] -> h2 f32[32][132]
    __shared__ __align__(16) unsigned short hs[32][264];  // H = relu(X W1 + b1), bf16
    __shared__ float red[256];
    __shared__ __align__(16) unsigned int w1b[2][4096];   // 2 x 16KB W1 K-slice
    __shared__ __align__(16) unsigned int w2b[2][2048];   // 2 x  8KB W2 K-slice

    unsigned short (*xs)[264] = (unsigned short (*)[264])xh_raw;
    float          (*h2)[132] = (float (*)[132])xh_raw;

    int tid = threadIdx.x;
    int R   = blockIdx.x * 32;
    const float* X = emb + (size_t)R * Dq;

    int w = tid >> 5, lane = tid & 31;
    int mrow = lane & 15, khalf = lane >> 4;
    union Op { v16bf v; unsigned int u[8]; uint4 q[2]; };

    // TDM: prefetch W1 K-slice 0 while we stage X
    if (w == 0)
        tdm_load_1d(wsu + WS_W1P, lds_off_of(&w1b[0][0]), 2048u, 16384u);

    for (int i = 0; i < 32; ++i) {           // stage + convert X
        int e = tid + i * 256;
        xs[e >> 8][e & 255] = f2bf(X[e]);
    }
    __syncthreads();

    // ---- GEMM1: [32,256]x[256,256], wave = 16x64 strip ----
    {
        int Mb = (w & 1) * 16;
        int Ns = w >> 1;
        Op A, Bv;
        v8f acc[4] = {};
        for (int kk = 0; kk < 8; ++kk) {
            if (w == 0) __builtin_amdgcn_s_wait_tensorcnt(0); // slice kk arrived
            __syncthreads();                                  // visible; prev buf free
            if (w == 0 && kk < 7)
                tdm_load_1d(wsu + WS_W1P + (kk + 1) * 4096,
                            lds_off_of(&w1b[(kk + 1) & 1][0]), 2048u, 16384u);

            const unsigned short* xr = &xs[Mb + mrow][0];
            int kb = kk * 32 + khalf * 8;
            A.u[0] = *(const unsigned int*)&xr[kb + 0];
            A.u[1] = *(const unsigned int*)&xr[kb + 2];
            A.u[2] = *(const unsigned int*)&xr[kb + 4];
            A.u[3] = *(const unsigned int*)&xr[kb + 6];
            A.u[4] = *(const unsigned int*)&xr[kb + 16];
            A.u[5] = *(const unsigned int*)&xr[kb + 18];
            A.u[6] = *(const unsigned int*)&xr[kb + 20];
            A.u[7] = *(const unsigned int*)&xr[kb + 22];
            const unsigned int* wsl = &w1b[kk & 1][0];
            for (int j = 0; j < 4; ++j) {
                int nt = Ns * 4 + j;
                const uint4* bp = (const uint4*)&wsl[(nt * 32 + lane) << 3];
                Bv.q[0] = bp[0]; Bv.q[1] = bp[1];
                acc[j] = __builtin_amdgcn_wmma_f32_16x16x32_bf16(
                    false, A.v, false, Bv.v, (short)0, acc[j], false, false);
            }
        }
        // prefetch W2 K-slice 0 while we write back H
        if (w == 0)
            tdm_load_1d(wsu + WS_W2P, lds_off_of(&w2b[0][0]), 1024u, 8192u);

        for (int j = 0; j < 4; ++j) {
            int col = (Ns * 4 + j) * 16 + mrow;
            float bias = wsf[WS_B1 + col];
            for (int r = 0; r < 8; ++r) {
                float val = acc[j][r] + bias;
                hs[Mb + r + 8 * khalf][col] = f2bf(val > 0.0f ? val : 0.0f);
            }
        }
    }
    __syncthreads();

    // ---- GEMM2: [32,256]x[256,128], wave = two 16x16 tiles ----
    {
        int Mb = (w & 1) * 16;
        Op A, Bv;
        v8f acc[2] = {};
        for (int kk = 0; kk < 8; ++kk) {
            if (w == 0) __builtin_amdgcn_s_wait_tensorcnt(0);
            __syncthreads();
            if (w == 0 && kk < 7)
                tdm_load_1d(wsu + WS_W2P + (kk + 1) * 2048,
                            lds_off_of(&w2b[(kk + 1) & 1][0]), 1024u, 8192u);

            const unsigned short* hr = &hs[Mb + mrow][0];
            int kb = kk * 32 + khalf * 8;
            A.u[0] = *(const unsigned int*)&hr[kb + 0];
            A.u[1] = *(const unsigned int*)&hr[kb + 2];
            A.u[2] = *(const unsigned int*)&hr[kb + 4];
            A.u[3] = *(const unsigned int*)&hr[kb + 6];
            A.u[4] = *(const unsigned int*)&hr[kb + 16];
            A.u[5] = *(const unsigned int*)&hr[kb + 18];
            A.u[6] = *(const unsigned int*)&hr[kb + 20];
            A.u[7] = *(const unsigned int*)&hr[kb + 22];
            const unsigned int* wsl = &w2b[kk & 1][0];
            for (int j = 0; j < 2; ++j) {
                int nt = (w >> 1) + 4 * j;
                const uint4* bp = (const uint4*)&wsl[(nt * 32 + lane) << 3];
                Bv.q[0] = bp[0]; Bv.q[1] = bp[1];
                acc[j] = __builtin_amdgcn_wmma_f32_16x16x32_bf16(
                    false, A.v, false, Bv.v, (short)0, acc[j], false, false);
            }
        }
        __syncthreads();    // xs fully dead on all waves before h2 aliases it
        for (int j = 0; j < 2; ++j) {
            int col = ((w >> 1) + 4 * j) * 16 + mrow;
            float bias = wsf[WS_B2 + col];
            for (int r = 0; r < 8; ++r) {
                float val = acc[j][r] + bias;
                h2[Mb + r + 8 * khalf][col] = val > 0.0f ? val : 0.0f;
            }
        }
    }
    __syncthreads();

    // ---- final dot + sigmoid + teacher blend + sigmoid ----
    int p = tid >> 3, c0 = (tid & 7) * 16;
    float part = 0.0f;
    for (int i = 0; i < 16; ++i) part += h2[p][c0 + i] * wsf[WS_W3 + c0 + i];
    red[tid] = part;
    __syncthreads();
    if ((tid & 7) == 0) {
        float sum = 0.0f;
        for (int i = 0; i < 8; ++i) sum += red[tid + i];
        sum += wsf[WS_SCAL + 1];
        float learned = 1.0f / (1.0f + __expf(-sum));
        int s = (R & (Sq - 1)) + p;
        int b = R >> 13;
        float comb = wsf[WS_TV + s] + wsf[WS_SCAL + 0] * learned;
        out[OUT_SC + b * Sq + s] = 1.0f / (1.0f + __expf(-comb));
    }
}

// ---------------------------------------------------------------------------
// Deterministic top-k per batch row: 2-level radix select on float bits,
// then prefix-scan compaction (no arrival-order atomics).
// ---------------------------------------------------------------------------
__global__ __launch_bounds__(1024) void topk_kernel(const float* __restrict__ sc_base,
                                                    int* __restrict__ wsidx,
                                                    float* __restrict__ outidx) {
    __shared__ unsigned int hist[256];
    __shared__ unsigned int scan[1024];
    __shared__ unsigned int sStat[4];
    int b = blockIdx.x, t = threadIdx.x;
    const float* sc = sc_base + b * Sq;

    unsigned int key16[8];
    for (int i = 0; i < 8; ++i) key16[i] = __float_as_uint(sc[t * 8 + i]) >> 16;

    if (t < 256) hist[t] = 0;
    __syncthreads();
    for (int i = 0; i < 8; ++i) atomicAdd(&hist[key16[i] >> 8], 1u);
    __syncthreads();
    if (t == 0) {
        unsigned acc = 0, C = 0, above = 0;
        for (int bin = 255; bin >= 0; --bin) {
            if (acc + hist[bin] >= (unsigned)Kq) { C = (unsigned)bin; above = acc; break; }
            acc += hist[bin];
        }
        sStat[0] = C; sStat[2] = above;
    }
    __syncthreads();
    unsigned C = sStat[0], above1 = sStat[2];
    if (t < 256) hist[t] = 0;
    __syncthreads();
    for (int i = 0; i < 8; ++i)
        if ((key16[i] >> 8) == C) atomicAdd(&hist[key16[i] & 255], 1u);
    __syncthreads();
    if (t == 0) {
        unsigned kp = (unsigned)Kq - above1;
        unsigned acc = 0, T2 = 0, above2 = 0;
        for (int sb = 255; sb >= 0; --sb) {
            if (acc + hist[sb] >= kp) { T2 = (unsigned)sb; above2 = acc; break; }
            acc += hist[sb];
        }
        unsigned nAb = above1 + above2;
        sStat[1] = (C << 8) | T2; sStat[2] = nAb; sStat[3] = (unsigned)Kq - nAb;
    }
    __syncthreads();
    unsigned T16 = sStat[1], nAb = sStat[2], kTie = sStat[3];

    unsigned cntA = 0, cntE = 0;
    for (int i = 0; i < 8; ++i) { cntA += (key16[i] > T16); cntE += (key16[i] == T16); }
    unsigned mine = (cntA << 16) | cntE;
    scan[t] = mine;
    __syncthreads();
    for (int off = 1; off < 1024; off <<= 1) {
        unsigned v = (t >= off) ? scan[t - off] : 0u;
        __syncthreads();
        scan[t] += v;
        __syncthreads();
    }
    unsigned ex = scan[t] - mine;
    unsigned exA = ex >> 16, exE = ex & 0xFFFFu;
    for (int i = 0; i < 8; ++i) {
        int s = t * 8 + i, pos = -1;
        if (key16[i] > T16)       { pos = (int)exA; exA++; }
        else if (key16[i] == T16) { if (exE < kTie) pos = (int)(nAb + exE); exE++; }
        if (pos >= 0) {
            wsidx[b * Kq + pos]  = s;
            outidx[b * Kq + pos] = (float)s;
        }
    }
}

__global__ void gather_kernel(const float* __restrict__ emb, const int* __restrict__ wsidx,
                              float* __restrict__ out) {
    int j = blockIdx.x;                 // b*Kq + r
    int b = j >> 11;
    int idx = wsidx[j];
    const float4* src = (const float4*)(emb + ((size_t)b * Sq + idx) * Dq);
    float4*       dst = (float4*)(out + (size_t)j * Dq);
    dst[threadIdx.x] = src[threadIdx.x];
}

extern "C" void kernel_launch(void* const* d_in, const int* in_sizes, int n_in,
                              void* d_out, int out_size, void* d_ws, size_t ws_size,
                              hipStream_t stream) {
    (void)in_sizes; (void)n_in; (void)out_size; (void)ws_size;
    const float* emb = (const float*)d_in[0];
    const float* W1  = (const float*)d_in[1];
    const float* b1  = (const float*)d_in[2];
    const float* g   = (const float*)d_in[3];
    const float* be  = (const float*)d_in[4];
    const float* mn  = (const float*)d_in[5];
    const float* vr  = (const float*)d_in[6];
    const float* W2  = (const float*)d_in[7];
    const float* b2  = (const float*)d_in[8];
    const float* W3  = (const float*)d_in[9];
    const float* b3  = (const float*)d_in[10];
    const float* tg  = (const float*)d_in[11];
    const float* tc  = (const float*)d_in[12];
    const float* gw  = (const float*)d_in[13];
    const float* cw  = (const float*)d_in[14];
    const float* tp  = (const float*)d_in[15];

    float*        out   = (float*)d_out;
    unsigned int* wsu   = (unsigned int*)d_ws;
    float*        wsf   = (float*)d_ws;
    int*          wsidx = (int*)(wsu + WS_TIDX);

    prep_weights<<<192, 256, 0, stream>>>(W1, W2, g, vr, wsu);
    prep_vec<<<32, 256, 0, stream>>>(b1, g, be, mn, vr, W2, b2, W3, b3, tg, tc, gw, cw, tp, wsf, out);
    scores_kernel<<<(Bq * Sq) / 32, 256, 0, stream>>>(emb, wsu, wsf, out);
    topk_kernel<<<Bq, 1024, 0, stream>>>(out + OUT_SC, wsidx, out + OUT_IDX);
    gather_kernel<<<Bq * Kq, 64, 0, stream>>>(emb, wsidx, out + OUT_SEL);
}
